// LSTM_695784702699
// MI455X (gfx1250) — matrix operand order repaired and checked by reading, not compiled, not run on hardware
//
#include <hip/hip_runtime.h>

typedef __attribute__((ext_vector_type(16))) _Float16 v16h;
typedef __attribute__((ext_vector_type(8)))  _Float16 v8h;
typedef __attribute__((ext_vector_type(8)))  float    v8f;

#define HSZ 32           // hidden size
#define WPB 4            // waves per block
#define RPW 16           // batch rows per wave (one WMMA M tile)
#define CT  32           // timesteps of x staged in LDS per chunk
#define HPAD 40          // padded h row stride in halves (80B -> bank spread)

#if __has_builtin(__builtin_amdgcn_tanhf)
#define HW_TANH 1
__device__ __forceinline__ float ftanh_(float x) {
    return __builtin_amdgcn_tanhf(x);
}
// y is ALREADY 0.5 * preactivation (folded into the weights):
// sigmoid(x) = 0.5 * tanh(0.5 x) + 0.5 = fma(0.5, tanh(y), 0.5)
__device__ __forceinline__ float fsigmoid_half(float y) {
    return __builtin_fmaf(0.5f, __builtin_amdgcn_tanhf(y), 0.5f);
}
#else
#define HW_TANH 0
__device__ __forceinline__ float ftanh_(float x) {
    float xc = fminf(fmaxf(x, -15.0f), 15.0f);
    float e  = __expf(-2.0f * xc);
    return (1.0f - e) * __builtin_amdgcn_rcpf(1.0f + e);
}
__device__ __forceinline__ float fsigmoid_half(float y) {
    // sigmoid(2y) = 1 / (1 + exp(-2y))
    return __builtin_amdgcn_rcpf(1.0f + __expf(-2.0f * y));
}
#endif

__global__ __launch_bounds__(WPB * 32)
void lstm_wmma_kernel(const float* __restrict__ x,      // [B, T]
                      const float* __restrict__ W_ih,   // [4H] (I==1)
                      const float* __restrict__ W_hh,   // [4H][H]
                      const float* __restrict__ b_ih,   // [4H]
                      const float* __restrict__ b_hh,   // [4H]
                      const float* __restrict__ W_lin,  // [H]
                      const float* __restrict__ b_lin,  // [1]
                      float* __restrict__ out,          // [B]
                      int Tlen)
{
    const int lane = threadIdx.x & 31;
    const int wave = threadIdx.x >> 5;
    const int hl   = lane >> 4;       // half-wave id (0/1)
    const int ln   = lane & 15;
    const int rowBase = (blockIdx.x * WPB + wave) * RPW;

    __shared__ __align__(16) _Float16 hsh[WPB][RPW][HPAD];
    __shared__ __align__(16) float    xsh[WPB][RPW][CT];

    // Per-lane gate-column constants (gate col of tile t for this lane = 16t+ln).
    // Tiles 0..3 (i,f) and 6..7 (o) are pre-scaled by 0.5 so sigmoid needs no
    // extra multiply; tiles 4..5 (g, plain tanh) stay at full scale.
    float wih_c[8], bs_c[8];
#pragma unroll
    for (int t = 0; t < 8; ++t) {
        int g = 16 * t + ln;
        float s = (t == 4 || t == 5) ? 1.0f : 0.5f;
        wih_c[t] = s * W_ih[g];
        bs_c[t]  = s * (b_ih[g] + b_hh[g]);
    }

    // B tiles = W_hh^T as 8 (32x16) f16 tiles, resident in VGPRs all kernel.
    // B k order used here (same as A): lane holds col n=ln; halves j=0..15 hold K=16*hl+j.
    // Same 0.5 pre-scale for the sigmoid gates (exact in f16: power of two).
    v16h Bt[8];
#pragma unroll
    for (int t = 0; t < 8; ++t) {
        const float* wrow = W_hh + (16 * t + ln) * HSZ + 16 * hl;  // contiguous K
        float s = (t == 4 || t == 5) ? 1.0f : 0.5f;
        v16h bb;
#pragma unroll
        for (int j = 0; j < 16; ++j) bb[j] = (_Float16)(s * wrow[j]);
        Bt[t] = bb;
    }

    // h0 = 0 in the LDS staging buffer.
    {
        _Float16* hz = &hsh[wave][0][0];
        for (int j = lane; j < RPW * HPAD; j += 32) hz[j] = (_Float16)0.0f;
    }

    // LSTM state: this lane owns rows M = r + 8*hl (r=0..7), cols {ln, ln+16}.
    float cst[8][2], hst[8][2];
#pragma unroll
    for (int r = 0; r < 8; ++r) {
        cst[r][0] = 0.f; cst[r][1] = 0.f; hst[r][0] = 0.f; hst[r][1] = 0.f;
    }

    for (int tb = 0; tb < Tlen; tb += CT) {
        // Stage x[rowBase..+15][tb..tb+CT) -> LDS (lane loads 16 floats of row ln).
        {
            const float* xr = x + (size_t)(rowBase + ln) * Tlen + tb + 16 * hl;
            float* xd = &xsh[wave][ln][16 * hl];
#pragma unroll
            for (int j = 0; j < 16; j += 4)
                *(float4*)(xd + j) = *(const float4*)(xr + j);
        }
        __syncthreads();

        for (int sc = 0; sc < CT; ++sc) {
            // A tile: h (16x32 f16). Lane row M=ln; halves 0..15 -> K=16*hl+j
            // (two b128 LDS loads).
            v16h A;
            {
                const _Float16* hr = &hsh[wave][ln][0];
                v8h lo = *(const v8h*)(hr + 16 * hl);
                v8h hi = *(const v8h*)(hr + 16 * hl + 8);
#pragma unroll
                for (int j = 0; j < 8; ++j) { A[j] = lo[j]; A[8 + j] = hi[j]; }
            }
            // x for this lane's 8 rows (broadcast LDS reads).
            float xv[8];
#pragma unroll
            for (int r = 0; r < 8; ++r) xv[r] = xsh[wave][8 * hl + r][sc];

#pragma unroll
            for (int hh = 0; hh < 2; ++hh) {   // hidden-col halves: tiles t = gate*2 + hh
                v8f Ci, Cf, Cg, Co;
#pragma unroll
                for (int r = 0; r < 8; ++r) {
                    Ci[r] = xv[r] * wih_c[0 + hh] + bs_c[0 + hh];
                    Cf[r] = xv[r] * wih_c[2 + hh] + bs_c[2 + hh];
                    Cg[r] = xv[r] * wih_c[4 + hh] + bs_c[4 + hh];
                    Co[r] = xv[r] * wih_c[6 + hh] + bs_c[6 + hh];
                }
                v8f Di = __builtin_amdgcn_wmma_f32_16x16x32_f16(false, A, false, Bt[0 + hh], (short)0, Ci, false, false);
                v8f Df = __builtin_amdgcn_wmma_f32_16x16x32_f16(false, A, false, Bt[2 + hh], (short)0, Cf, false, false);
                v8f Dg = __builtin_amdgcn_wmma_f32_16x16x32_f16(false, A, false, Bt[4 + hh], (short)0, Cg, false, false);
                v8f Do = __builtin_amdgcn_wmma_f32_16x16x32_f16(false, A, false, Bt[6 + hh], (short)0, Co, false, false);
#pragma unroll
                for (int r = 0; r < 8; ++r) {
                    float ig = fsigmoid_half(Di[r]);   // D already 0.5*preact
                    float fg = fsigmoid_half(Df[r]);
                    float gg = ftanh_(Dg[r]);
                    float og = fsigmoid_half(Do[r]);
                    float c  = fg * cst[r][hh] + ig * gg;
                    cst[r][hh] = c;
                    hst[r][hh] = og * ftanh_(c);
                }
            }

            // Write h back (f16, row-major) for next step's A load.
#pragma unroll
            for (int r = 0; r < 8; ++r) {
                hsh[wave][8 * hl + r][ln]      = (_Float16)hst[r][0];
                hsh[wave][8 * hl + r][ln + 16] = (_Float16)hst[r][1];
            }
        }
        __syncthreads();
    }

    // out[M] = sum_n h[M][n] * W_lin[n] + b_lin ; reduce within each half-wave.
    float wl0 = W_lin[ln], wl1 = W_lin[ln + 16];
    float blin = b_lin[0];
#pragma unroll
    for (int r = 0; r < 8; ++r) {
        float p = hst[r][0] * wl0 + hst[r][1] * wl1;
        p += __shfl_xor(p, 1, 32);
        p += __shfl_xor(p, 2, 32);
        p += __shfl_xor(p, 4, 32);
        p += __shfl_xor(p, 8, 32);
        if (ln == 0) out[rowBase + 8 * hl + r] = p + blin;
    }
}

extern "C" void kernel_launch(void* const* d_in, const int* in_sizes, int n_in,
                              void* d_out, int out_size, void* d_ws, size_t ws_size,
                              hipStream_t stream) {
    const float* x     = (const float*)d_in[0];
    const float* W_ih  = (const float*)d_in[1];
    const float* W_hh  = (const float*)d_in[2];
    const float* b_ih  = (const float*)d_in[3];
    const float* b_hh  = (const float*)d_in[4];
    const float* W_lin = (const float*)d_in[5];
    const float* b_lin = (const float*)d_in[6];
    float* out = (float*)d_out;

    int Bsz  = out_size;               // 8192
    int Tlen = in_sizes[0] / Bsz;      // 512 (x is [B,T,1])
    int blocks = Bsz / (WPB * RPW);    // 64 rows per block -> 128 blocks

    lstm_wmma_kernel<<<blocks, WPB * 32, 0, stream>>>(
        x, W_ih, W_hh, b_ih, b_hh, W_lin, b_lin, out, Tlen);
}
